// Generator_Model_59219009077300
// MI455X (gfx1250) — matrix operand order repaired and checked by reading, not compiled, not run on hardware
//
#include <hip/hip_runtime.h>
#include <hip/hip_bf16.h>

#define B_  32
#define T_  128
#define E_  100
#define H_  700
#define DY_ 200
#define DZ_ 500
#define V_  32000
#define H3  2100          // 3*H
#define KP  704           // K padded to multiple of 32 (22*32)
#define NK  22            // KP/32 k-steps
#define BT  4096          // B*T
#define GP  2112          // gate-padded recurrent width: 3*704
#define TILE_HW 11264     // halfwords per (16-col x 704-K) fragment-swizzled tile

typedef __attribute__((ext_vector_type(16))) __bf16 v16bf;
typedef __attribute__((ext_vector_type(8)))  float  v8f;

__device__ __forceinline__ float sigmoidf_(float x) { return 1.0f / (1.0f + __expf(-x)); }

// float <-> bf16 via bit manipulation (RNE)
__device__ __forceinline__ __bf16 tobf(float f) {
    unsigned int x = __float_as_uint(f);
    unsigned int r = (x + 0x7FFFu + ((x >> 16) & 1u)) >> 16;
    unsigned short s = (unsigned short)r;
    __bf16 h; __builtin_memcpy(&h, &s, 2);
    return h;
}
__device__ __forceinline__ float bf2f(__bf16 h) {
    unsigned short s; __builtin_memcpy(&s, &h, 2);
    return __uint_as_float((unsigned int)s << 16);
}

__device__ __forceinline__ v8f wmma_bf16(v16bf a, v16bf b, v8f c) {
    return __builtin_amdgcn_wmma_f32_16x16x32_bf16(false, a, false, b, (short)0, c, false, false);
}

union Frag { v16bf v; uint4 q[2]; };

// A fragment (row-major bf16 matrix with row stride KP):
//   lane<16 : K 0..15 of its row
//   lane>=16: K 16..31
__device__ __forceinline__ void loadA(Frag& f, const __bf16* A, int row, int kk, int hi) {
    const __bf16* p = A + (size_t)row * KP + kk * 32;
    f.q[0] = *(const uint4*)(p + hi * 16);
    f.q[1] = *(const uint4*)(p + hi * 16 + 8);
}
// B fragment from fragment-swizzled global layout:
//   element (k, n): tile = n>>4; halfword addr = tile*11264 + (k>>5)*512 + (n&15)*32 + (k&31)
//   -> lane ml owns column n (K 0..15 for lanes<16, K 16..31 for lanes>=16), two contiguous b128 loads
__device__ __forceinline__ void loadB(Frag& f, const __bf16* B, int tile, int kk, int ml, int hi) {
    const __bf16* p = B + (size_t)tile * TILE_HW + (size_t)kk * 512 + ml * 32 + hi * 16;
    f.q[0] = *(const uint4*)(p);
    f.q[1] = *(const uint4*)(p + 8);
}

// ---------------- hA[0] = bf16([labels @ W1 + b1 || z]), pad cols zeroed ----------------
__global__ void k_init_h0(const float* __restrict__ labels, const float* __restrict__ z,
                          const float* __restrict__ W1, const float* __restrict__ b1,
                          __bf16* __restrict__ hA0) {
    int i = blockIdx.x * blockDim.x + threadIdx.x;
    if (i >= B_ * KP) return;
    int b = i / KP, j = i % KP;
    float v = 0.0f;
    if (j < DY_)     v = labels[b] * W1[j] + b1[j];
    else if (j < H_) v = z[(size_t)b * DZ_ + (j - DY_)];
    hA0[i] = tobf(v);
}

// ---------------- xp[row, :] = emb[dec[row]] @ gru1_k + gru1_b[0] ----------------
__global__ void k_xp(const int* __restrict__ dec, const float* __restrict__ emb,
                     const float* __restrict__ k1, const float* __restrict__ gb,
                     float* __restrict__ xp) {
    __shared__ float se[E_];
    int row = blockIdx.x;          // 0..4095 == (b*T + t)
    int tid = threadIdx.x;
    int tok = dec[row];
    for (int e = tid; e < E_; e += blockDim.x) se[e] = emb[(size_t)tok * E_ + e];
    __syncthreads();
    for (int j = tid; j < H3; j += blockDim.x) {
        float acc = gb[j];
        #pragma unroll 4
        for (int e = 0; e < E_; ++e) acc += se[e] * k1[(size_t)e * H3 + j];
        xp[(size_t)row * H3 + j] = acc;
    }
}

// ---------------- rk -> bf16, gate-padded (3 x 704 cols), fragment-swizzled ----------------
__global__ void k_pack_rk(const float* __restrict__ rk, __bf16* __restrict__ rkB) {
    size_t d = (size_t)blockIdx.x * blockDim.x + threadIdx.x;
    if (d >= (size_t)KP * GP) return;
    int tile = (int)(d / TILE_HW);
    int r    = (int)(d % TILE_HW);
    int kk = r >> 9, r2 = r & 511, ml = r2 >> 5, kin = r2 & 31;
    int n = tile * 16 + ml;        // gate-padded column 0..2111
    int k = kk * 32 + kin;
    int g = n / KP, j = n % KP;
    float v = (k < H_ && j < H_) ? rk[(size_t)k * H3 + g * H_ + j] : 0.0f;
    rkB[d] = tobf(v);
}

// ---------------- proj_W -> bf16, K padded to 704, fragment-swizzled ----------------
__global__ void k_pack_W(const float* __restrict__ W, __bf16* __restrict__ Wb) {
    size_t d = (size_t)blockIdx.x * blockDim.x + threadIdx.x;
    if (d >= (size_t)KP * V_) return;
    int tile = (int)(d / TILE_HW);
    int r    = (int)(d % TILE_HW);
    int kk = r >> 9, r2 = r & 511, ml = r2 >> 5, kin = r2 & 31;
    int n = tile * 16 + ml;
    int k = kk * 32 + kin;
    float v = (k < H_) ? W[(size_t)k * V_ + n] : 0.0f;
    Wb[d] = tobf(v);
}

// ---------------- one GRU step as WMMA GEMM: hp = hA @ rkB (all 3 gates), gates fused ----------------
// 11 blocks x 128 threads = 44 waves; wave nt owns 16 columns across z/r/h gate tiles.
__global__ __launch_bounds__(128) void k_gru_step(
        const float* __restrict__ xp, const __bf16* __restrict__ rkB,
        const float* __restrict__ gb, const __bf16* __restrict__ hAin,
        __bf16* __restrict__ hAout, __bf16* __restrict__ gA,
        const float* __restrict__ mask, int step) {
    const int lane = threadIdx.x & 31;
    const int nt = blockIdx.x * 4 + (threadIdx.x >> 5);   // 0..43 column tile
    const int hi = lane >> 4, ml = lane & 15;

    v8f acc[3][2] = {};   // [gate][m-tile of 16 batch rows]
    for (int kk = 0; kk < NK; ++kk) {
        Frag af[2], bf;
        loadA(af[0], hAin, ml,      kk, hi);
        loadA(af[1], hAin, 16 + ml, kk, hi);
        #pragma unroll
        for (int g = 0; g < 3; ++g) {
            loadB(bf, rkB, g * 44 + nt, kk, ml, hi);
            acc[g][0] = wmma_bf16(af[0].v, bf.v, acc[g][0]);
            acc[g][1] = wmma_bf16(af[1].v, bf.v, acc[g][1]);
        }
    }

    int c = nt * 16 + ml;
    #pragma unroll
    for (int tm = 0; tm < 2; ++tm) {
        #pragma unroll
        for (int i = 0; i < 8; ++i) {
            int b = tm * 16 + hi * 8 + i;            // D layout: VGPR i -> M = i (+8 for hi lanes)
            size_t row = (size_t)b * T_ + step;
            if (c < H_) {
                float az = acc[0][tm][i] + gb[H3 + c];
                float ar = acc[1][tm][i] + gb[H3 + H_ + c];
                float ah = acc[2][tm][i] + gb[H3 + 2 * H_ + c];
                size_t xo = row * H3;
                float zg = sigmoidf_(xp[xo + c] + az);
                float rg = sigmoidf_(xp[xo + H_ + c] + ar);
                float hh = tanhf(xp[xo + 2 * H_ + c] + rg * ah);
                float hp = bf2f(hAin[(size_t)b * KP + c]);
                float hn = zg * hp + (1.0f - zg) * hh;
                hAout[(size_t)b * KP + c] = tobf(hn);
                gA[row * KP + c] = tobf(hn * mask[row * H_ + c]);
            } else {
                hAout[(size_t)b * KP + c] = tobf(0.0f);  // keep K-pad zero
                gA[row * KP + c] = tobf(0.0f);
            }
        }
    }
}

// ---------------- projection: [4096 x 704]bf16 @ [704 x 32000]bf16 + bias, pure global/L2 + WMMA ----
// 512 threads (16 waves), 256x256 tile per block; wave = 2 m-tiles x 8 n-tiles, no LDS, no barriers.
__global__ __launch_bounds__(512) void k_proj(const __bf16* __restrict__ gA,
                                              const __bf16* __restrict__ Wb,
                                              const float* __restrict__ bias,
                                              float* __restrict__ out) {
    const int tid  = threadIdx.x;
    const int lane = tid & 31;
    const int wave = tid >> 5;            // 0..15
    const int mg = wave >> 1;             // 0..7  -> 32-row slab
    const int ng = wave & 1;              // 0..1  -> 8 n-tile group
    const int hi = lane >> 4, ml = lane & 15;
    const int rowBase = blockIdx.y * 256;
    const int ntBase  = blockIdx.x * 16 + ng * 8;    // global 16-col tile index

    v8f acc[2][8] = {};
    for (int kk = 0; kk < NK; ++kk) {
        Frag af[2], bf;
        loadA(af[0], gA, rowBase + mg * 32 + ml,      kk, hi);
        loadA(af[1], gA, rowBase + mg * 32 + 16 + ml, kk, hi);
        #pragma unroll
        for (int tn = 0; tn < 8; ++tn) {
            loadB(bf, Wb, ntBase + tn, kk, ml, hi);
            acc[0][tn] = wmma_bf16(af[0].v, bf.v, acc[0][tn]);
            acc[1][tn] = wmma_bf16(af[1].v, bf.v, acc[1][tn]);
        }
    }
    #pragma unroll
    for (int tm = 0; tm < 2; ++tm) {
        #pragma unroll
        for (int tn = 0; tn < 8; ++tn) {
            int col = (ntBase + tn) * 16 + ml;
            float bv = bias[col];
            #pragma unroll
            for (int i = 0; i < 8; ++i) {
                int row = rowBase + mg * 32 + tm * 16 + hi * 8 + i;
                out[(size_t)row * V_ + col] = acc[tm][tn][i] + bv;
            }
        }
    }
}

extern "C" void kernel_launch(void* const* d_in, const int* in_sizes, int n_in,
                              void* d_out, int out_size, void* d_ws, size_t ws_size,
                              hipStream_t stream) {
    (void)in_sizes; (void)n_in; (void)out_size; (void)ws_size;
    const float* labels = (const float*)d_in[0];
    const float* z      = (const float*)d_in[1];
    const int*   dec    = (const int*)d_in[2];
    const float* mask   = (const float*)d_in[3];
    const float* emb    = (const float*)d_in[4];
    const float* W1     = (const float*)d_in[5];
    const float* b1     = (const float*)d_in[6];
    const float* g1k    = (const float*)d_in[7];
    const float* g1rk   = (const float*)d_in[8];
    const float* g1b    = (const float*)d_in[9];
    const float* pW     = (const float*)d_in[10];
    const float* pb     = (const float*)d_in[11];
    // d_in[12..14] = gru3_* : dead code in the reference, skipped.
    float* out = (float*)d_out;

    char* ws = (char*)d_ws;
    size_t off = 0;
    auto wsalloc = [&](size_t bytes) -> void* {
        void* p = ws + off;
        off = (off + bytes + 255) & ~(size_t)255;
        return p;
    };
    float*  xp  = (float*) wsalloc((size_t)BT * H3 * sizeof(float));  // 34.4 MB
    __bf16* hA0 = (__bf16*)wsalloc((size_t)B_ * KP * 2);              // 45 KB (ping)
    __bf16* hA1 = (__bf16*)wsalloc((size_t)B_ * KP * 2);              // 45 KB (pong)
    __bf16* gA  = (__bf16*)wsalloc((size_t)BT * KP * 2);              // 5.8 MB
    __bf16* rkB = (__bf16*)wsalloc((size_t)KP * GP * 2);              // 3.0 MB
    __bf16* Wb  = (__bf16*)wsalloc((size_t)KP * V_ * 2);              // 45 MB

    k_init_h0<<<(B_ * KP + 255) / 256, 256, 0, stream>>>(labels, z, W1, b1, hA0);
    k_xp<<<BT, 256, 0, stream>>>(dec, emb, g1k, g1b, xp);
    k_pack_rk<<<(int)(((size_t)KP * GP + 255) / 256), 256, 0, stream>>>(g1rk, rkB);
    k_pack_W<<<(int)(((size_t)KP * V_ + 255) / 256), 256, 0, stream>>>(pW, Wb);

    __bf16* hp[2] = { hA0, hA1 };
    for (int t = 0; t < T_; ++t)
        k_gru_step<<<11, 128, 0, stream>>>(xp, rkB, g1b, hp[t & 1], hp[(t + 1) & 1],
                                           gA, mask, t);

    k_proj<<<dim3(V_ / 256, BT / 256), 512, 0, stream>>>(gA, Wb, pb, out);
}
